// GCNSim_47665547051796
// MI455X (gfx1250) — matrix operand order repaired and checked by reading, not compiled, not run on hardware
//
#include <hip/hip_runtime.h>
#include <hip/hip_bf16.h>

// ---------------- problem constants (match reference) ----------------
#define N_NODES   100000
#define IN_FEATS  256
#define N_EDGES   3200000
#define N_SEL     4096
#define J_POOL    50
#define M_PAIRS   (N_SEL * J_POOL)      // 204800
#define H1_F      32
#define H2_F      16
#define ROW_TILES (N_NODES / 16)        // 6250 exactly

typedef __attribute__((ext_vector_type(16))) __bf16 v16bf;
typedef __attribute__((ext_vector_type(8)))  float  v8f;

// Native f32 -> bf16 convert (RNE); lowers to v_cvt_pk_bf16_f32 on gfx1250
static __device__ __forceinline__ __bf16 f2bf(float f) {
    return (__bf16)f;
}

// ---------------- degree / normalization ----------------
__global__ __launch_bounds__(256) void k_deg_init(float* deg) {
    int n = blockIdx.x * 256 + threadIdx.x;
    if (n < N_NODES) deg[n] = 1.0f;                 // self-loop
}

__global__ __launch_bounds__(256) void k_deg_edges(const int* __restrict__ dst, float* deg) {
    int e = blockIdx.x * 256 + threadIdx.x;
    if (e < N_EDGES) atomicAdd(&deg[dst[e]], 1.0f);
}

__global__ __launch_bounds__(256) void k_dis(float* deg) {
    int n = blockIdx.x * 256 + threadIdx.x;
    if (n < N_NODES) deg[n] = rsqrtf(fmaxf(deg[n], 1e-12f));
}

// ---------------- GEMM1: hw1 = x(100000x256) @ w1(256x32), bf16 WMMA ----------------
// LDS holds w1 pre-swizzled into the ISA B-operand layout:
//   sw[((kc*2 + ct)*32 + lane)*16 + i] = w1[(kc*32 + kbB(lane) + i + (i&8))*32 + ct*16 + (lane&15)]
// so each lane reads its 16 B halfs as one contiguous 32-byte LDS access per K-step.
__global__ __launch_bounds__(256) void k_gemm1(const float* __restrict__ x,
                                               const float* __restrict__ w1,
                                               float* __restrict__ hw1) {
    __shared__ __attribute__((aligned(32))) __bf16 sw[IN_FEATS * H1_F]; // 16KB
    for (int idx = threadIdx.x; idx < IN_FEATS * H1_F; idx += 256) {
        int i   = idx & 15;
        int ln  = (idx >> 4) & 31;
        int ctk = idx >> 9;              // kc*2 + ct
        int ct  = ctk & 1;
        int kc  = ctk >> 1;
        int kb  = (ln & 16) ? 8 : 0;
        int col = ct * 16 + (ln & 15);
        sw[idx] = f2bf(w1[(kc * 32 + kb + i + (i & 8)) * H1_F + col]);
    }
    __syncthreads();

    const int wave = threadIdx.x >> 5;
    const int lane = threadIdx.x & 31;
    const int rowtile = blockIdx.x * 4 + (wave >> 1);
    const int coltile = wave & 1;
    if (rowtile >= ROW_TILES) return;               // uniform per wave: EXEC all-1s

    const int row = rowtile * 16 + (lane & 15);
    const int col = coltile * 16 + (lane & 15);
    const int kbA = (lane & 16) ? 8 : 0;            // ISA A-layout (16-bit 16x32)

    v8f acc = {};
    const float* xr = x + row * IN_FEATS;
#pragma unroll
    for (int kk = 0; kk < IN_FEATS; kk += 32) {
        v16bf a;
        const float* pa = xr + kk + kbA;
#pragma unroll
        for (int i = 0; i < 8; ++i) a[i]     = f2bf(pa[i]);
#pragma unroll
        for (int i = 0; i < 8; ++i) a[8 + i] = f2bf(pa[16 + i]);
        // one contiguous 32B LDS read per lane: the pre-swizzled B fragment
        v16bf b = *(const v16bf*)(sw + ((((kk >> 5) << 1) + coltile) << 9) + lane * 16);
        acc = __builtin_amdgcn_wmma_f32_16x16x32_bf16(false, a, false, b,
                                                      (short)0, acc, false, false);
    }
    float* out = hw1 + (rowtile * 16 + 8 * (lane >> 4)) * H1_F + col;
#pragma unroll
    for (int r = 0; r < 8; ++r) out[r * H1_F] = acc[r];
}

// ---------------- layer-1 aggregate ----------------
__global__ __launch_bounds__(256) void k_l1_init(const float* __restrict__ hw1,
                                                 const float* __restrict__ dis,
                                                 float* __restrict__ h1) {
    int idx = blockIdx.x * 256 + threadIdx.x;       // N_NODES*32 threads
    if (idx >= N_NODES * H1_F) return;
    int n = idx >> 5;
    float d = dis[n];
    h1[idx] = hw1[idx] * d * d;                     // self-loop contribution
}

__global__ __launch_bounds__(256) void k_l1_edges(const int* __restrict__ src,
                                                  const int* __restrict__ dst,
                                                  const float* __restrict__ dis,
                                                  const float* __restrict__ hw1,
                                                  float* __restrict__ h1) {
    int e = blockIdx.x * 256 + threadIdx.x;
    if (e >= N_EDGES) return;
    int s = src[e], d = dst[e];
    float w = dis[s] * dis[d];
    const float4* p = (const float4*)(hw1 + s * H1_F);
    float* q = h1 + d * H1_F;
#pragma unroll
    for (int i = 0; i < 8; ++i) {
        float4 v = p[i];
        atomicAdd(q + i * 4 + 0, v.x * w);
        atomicAdd(q + i * 4 + 1, v.y * w);
        atomicAdd(q + i * 4 + 2, v.z * w);
        atomicAdd(q + i * 4 + 3, v.w * w);
    }
}

__global__ __launch_bounds__(256) void k_l1_post(float* h1, const float* __restrict__ b1) {
    int idx = blockIdx.x * 256 + threadIdx.x;
    if (idx >= N_NODES * H1_F) return;
    h1[idx] = fmaxf(h1[idx] + b1[idx & 31], 0.0f);  // bias + ReLU
}

// ---------------- GEMM2: hw2 = h1(100000x32) @ w2(32x16), one WMMA per tile ----------------
__global__ __launch_bounds__(256) void k_gemm2(const float* __restrict__ h1,
                                               const float* __restrict__ w2,
                                               float* __restrict__ hw2) {
    __shared__ __attribute__((aligned(32))) __bf16 sw[H1_F * H2_F];   // pre-swizzled B
    for (int idx = threadIdx.x; idx < H1_F * H2_F; idx += 256) {
        int i   = idx & 15;
        int ln  = idx >> 4;
        int kb  = (ln & 16) ? 8 : 0;
        int col = ln & 15;
        sw[idx] = f2bf(w2[(kb + i + (i & 8)) * H2_F + col]);
    }
    __syncthreads();

    const int wave = threadIdx.x >> 5;
    const int lane = threadIdx.x & 31;
    const int rowtile = blockIdx.x * 8 + wave;
    if (rowtile >= ROW_TILES) return;

    const int row = rowtile * 16 + (lane & 15);
    const int col = lane & 15;
    const int kbA = (lane & 16) ? 8 : 0;

    v16bf a;
    const float* pa = h1 + row * H1_F + kbA;
#pragma unroll
    for (int i = 0; i < 8; ++i) a[i]     = f2bf(pa[i]);
#pragma unroll
    for (int i = 0; i < 8; ++i) a[8 + i] = f2bf(pa[16 + i]);
    v16bf b = *(const v16bf*)(sw + lane * 16);

    v8f acc = {};
    acc = __builtin_amdgcn_wmma_f32_16x16x32_bf16(false, a, false, b,
                                                  (short)0, acc, false, false);
    float* out = hw2 + (rowtile * 16 + 8 * (lane >> 4)) * H2_F + col;
#pragma unroll
    for (int r = 0; r < 8; ++r) out[r * H2_F] = acc[r];
}

// ---------------- layer-2 aggregate ----------------
__global__ __launch_bounds__(256) void k_l2_init(const float* __restrict__ hw2,
                                                 const float* __restrict__ dis,
                                                 const float* __restrict__ b2,
                                                 float* __restrict__ h2) {
    int idx = blockIdx.x * 256 + threadIdx.x;       // N_NODES*16 threads
    if (idx >= N_NODES * H2_F) return;
    int n = idx >> 4;
    float d = dis[n];
    h2[idx] = hw2[idx] * d * d + b2[idx & 15];      // self-loop + bias (no relu)
}

__global__ __launch_bounds__(256) void k_l2_edges(const int* __restrict__ src,
                                                  const int* __restrict__ dst,
                                                  const float* __restrict__ dis,
                                                  const float* __restrict__ hw2,
                                                  float* __restrict__ h2) {
    int e = blockIdx.x * 256 + threadIdx.x;
    if (e >= N_EDGES) return;
    int s = src[e], d = dst[e];
    float w = dis[s] * dis[d];
    const float4* p = (const float4*)(hw2 + s * H2_F);
    float* q = h2 + d * H2_F;
#pragma unroll
    for (int i = 0; i < 4; ++i) {
        float4 v = p[i];
        atomicAdd(q + i * 4 + 0, v.x * w);
        atomicAdd(q + i * 4 + 1, v.y * w);
        atomicAdd(q + i * 4 + 2, v.z * w);
        atomicAdd(q + i * 4 + 3, v.w * w);
    }
}

// ---------------- pairwise concat + MoE head + log_softmax ----------------
__global__ __launch_bounds__(256) void k_moe(const float* __restrict__ h2,
                                             const int* __restrict__ nodes,
                                             const float* __restrict__ g1w,
                                             const float* __restrict__ e1w,
                                             const float* __restrict__ e1b,
                                             const float* __restrict__ g2w,
                                             const float* __restrict__ e2w,
                                             const float* __restrict__ e2b,
                                             float* __restrict__ out_lsm,
                                             float* __restrict__ out_xs) {
    __shared__ float s_g1[32 * 4];
    __shared__ float s_e1w[4 * 32 * 16];
    __shared__ float s_e1b[4 * 16];
    __shared__ float s_g2[16 * 4];
    __shared__ float s_e2w[4 * 16 * 2];
    __shared__ float s_e2b[4 * 2];
    for (int i = threadIdx.x; i < 32 * 4;      i += 256) s_g1[i]  = g1w[i];
    for (int i = threadIdx.x; i < 4 * 32 * 16; i += 256) s_e1w[i] = e1w[i];
    for (int i = threadIdx.x; i < 4 * 16;      i += 256) s_e1b[i] = e1b[i];
    for (int i = threadIdx.x; i < 16 * 4;      i += 256) s_g2[i]  = g2w[i];
    for (int i = threadIdx.x; i < 4 * 16 * 2;  i += 256) s_e2w[i] = e2w[i];
    for (int i = threadIdx.x; i < 4 * 2;       i += 256) s_e2b[i] = e2b[i];
    __syncthreads();

    int m = blockIdx.x * 256 + threadIdx.x;         // grid sized exactly M_PAIRS
    int ni = nodes[m / J_POOL];
    int nj = nodes[m % J_POOL];

    float xs[32];
#pragma unroll
    for (int f = 0; f < 16; ++f) xs[f]      = h2[ni * H2_F + f];
#pragma unroll
    for (int f = 0; f < 16; ++f) xs[16 + f] = h2[nj * H2_F + f];
    float* xo = out_xs + m * 32;
#pragma unroll
    for (int f = 0; f < 32; ++f) xo[f] = xs[f];

    // gate 1 (argmax = first max, matching jnp.argmax)
    float gl[4] = {0.f, 0.f, 0.f, 0.f};
#pragma unroll
    for (int k = 0; k < 32; ++k) {
        float v = xs[k];
#pragma unroll
        for (int e2 = 0; e2 < 4; ++e2) gl[e2] += v * s_g1[k * 4 + e2];
    }
    int b1i = 0;
#pragma unroll
    for (int e2 = 1; e2 < 4; ++e2) if (gl[e2] > gl[b1i]) b1i = e2;

    // expert 1 + ReLU
    float z1[16];
#pragma unroll
    for (int o = 0; o < 16; ++o) {
        float v = s_e1b[b1i * 16 + o];
#pragma unroll
        for (int k = 0; k < 32; ++k) v += xs[k] * s_e1w[b1i * 512 + k * 16 + o];
        z1[o] = fmaxf(v, 0.0f);
    }

    // gate 2
    float g2l[4] = {0.f, 0.f, 0.f, 0.f};
#pragma unroll
    for (int k = 0; k < 16; ++k) {
        float v = z1[k];
#pragma unroll
        for (int e2 = 0; e2 < 4; ++e2) g2l[e2] += v * s_g2[k * 4 + e2];
    }
    int b2i = 0;
#pragma unroll
    for (int e2 = 1; e2 < 4; ++e2) if (g2l[e2] > g2l[b2i]) b2i = e2;

    // expert 2
    float z2[2];
#pragma unroll
    for (int o = 0; o < 2; ++o) {
        float v = s_e2b[b2i * 2 + o];
#pragma unroll
        for (int k = 0; k < 16; ++k) v += z1[k] * s_e2w[b2i * 32 + k * 2 + o];
        z2[o] = v;
    }

    // log_softmax over 2 classes
    float mx  = fmaxf(z2[0], z2[1]);
    float lse = mx + logf(expf(z2[0] - mx) + expf(z2[1] - mx));
    out_lsm[m * 2 + 0] = z2[0] - lse;
    out_lsm[m * 2 + 1] = z2[1] - lse;
}

// ---------------- launcher ----------------
extern "C" void kernel_launch(void* const* d_in, const int* in_sizes, int n_in,
                              void* d_out, int out_size, void* d_ws, size_t ws_size,
                              hipStream_t stream) {
    const float* x     = (const float*)d_in[0];
    const float* w1    = (const float*)d_in[1];
    const float* b1    = (const float*)d_in[2];
    const float* w2    = (const float*)d_in[3];
    const float* b2    = (const float*)d_in[4];
    const float* g1w   = (const float*)d_in[5];
    const float* e1w   = (const float*)d_in[6];
    const float* e1b   = (const float*)d_in[7];
    const float* g2w   = (const float*)d_in[8];
    const float* e2w   = (const float*)d_in[9];
    const float* e2b   = (const float*)d_in[10];
    const int*   eidx  = (const int*)d_in[11];
    const int*   nodes = (const int*)d_in[12];
    const int* src = eidx;
    const int* dst = eidx + N_EDGES;

    float* dis  = (float*)d_ws;                       // 100000
    float* bufA = dis  + N_NODES;                     // 3.2M  (hw1, later hw2)
    float* bufB = bufA + N_NODES * H1_F;              // 3.2M  (h1,  later h2)
    float* hw1 = bufA;
    float* h1  = bufB;
    float* hw2 = bufA;                                // aliases hw1 (dead by then)
    float* h2  = bufB;                                // aliases h1  (dead by then)

    float* out_lsm = (float*)d_out;                   // [M_PAIRS, 2]
    float* out_xs  = out_lsm + M_PAIRS * 2;           // [M_PAIRS, 32]

    const int nodeBlks = (N_NODES + 255) / 256;       // 391
    const int edgeBlks = (N_EDGES + 255) / 256;       // 12500

    k_deg_init <<<nodeBlks, 256, 0, stream>>>(dis);
    k_deg_edges<<<edgeBlks, 256, 0, stream>>>(dst, dis);
    k_dis      <<<nodeBlks, 256, 0, stream>>>(dis);

    k_gemm1    <<<(ROW_TILES + 3) / 4, 256, 0, stream>>>(x, w1, hw1);
    k_l1_init  <<<(N_NODES * H1_F + 255) / 256, 256, 0, stream>>>(hw1, dis, h1);
    k_l1_edges <<<edgeBlks, 256, 0, stream>>>(src, dst, dis, hw1, h1);
    k_l1_post  <<<(N_NODES * H1_F + 255) / 256, 256, 0, stream>>>(h1, b1);

    k_gemm2    <<<(ROW_TILES + 7) / 8, 256, 0, stream>>>(h1, w2, hw2);
    k_l2_init  <<<(N_NODES * H2_F + 255) / 256, 256, 0, stream>>>(hw2, dis, b2, h2);
    k_l2_edges <<<edgeBlks, 256, 0, stream>>>(src, dst, dis, hw2, h2);

    k_moe      <<<M_PAIRS / 256, 256, 0, stream>>>(h2, nodes, g1w, e1w, e1b,
                                                   g2w, e2w, e2b, out_lsm, out_xs);
}
